// SS2D_13048110645923
// MI455X (gfx1250) — matrix operand order repaired and checked by reading, not compiled, not run on hardware
//
#include <hip/hip_runtime.h>
#include <hip/hip_bf16.h>
#include <math.h>

// SS2D cross-modal selective scan for MI455X (gfx1250, wave32).
// Pipeline: [zero ymerge] -> [WMMA proj dbl=(dt|B|C)] -> [WMMA dt-proj + softplus]
//           -> [sequential scan, atomicAdd cross-merge] -> [layernorm -> out]

#define B_  2
#define D_  192
#define HW_ 56
#define N_  16
#define R_  12
#define K_  4
#define L_  3136      // 56*56
#define CC_ 44        // R + 2N

typedef __attribute__((ext_vector_type(16))) _Float16 v16h;
typedef __attribute__((ext_vector_type(8)))  float    v8f;

__device__ __forceinline__ int scan_idx(int k, int l) {
    // spatial index for scan-order k at scan position l (== merge position)
    if (k == 0) return l;
    if (k == 1) return (l % HW_) * HW_ + (l / HW_);
    if (k == 2) return (L_ - 1) - l;
    int lr = (L_ - 1) - l;
    return (lr % HW_) * HW_ + (lr / HW_);
}

// ---------------------------------------------------------------------------
// Zero the merge accumulators (d_ws is poisoned; we atomicAdd into these).
__global__ __launch_bounds__(256) void k_zero(float* __restrict__ p, int n) {
    int i = blockIdx.x * 256 + threadIdx.x;
    if (i < n) p[i] = 0.f;
}

// ---------------------------------------------------------------------------
// GEMM1: dbl[b,k,c,l] = sum_d W[k,c,d] * Xs[b,k,d,l], Xs gathered on the fly.
// One wave computes a 16x16 (c x l) tile via 6 WMMA steps over D=192.
// All fragment loads are unconditional (clamped addresses + result masking)
// so no exec-mask branches appear in the inner loop.
__global__ __launch_bounds__(128) void k_proj_dbl(
    const float* __restrict__ x1, const float* __restrict__ x2,
    const float* __restrict__ w1, const float* __restrict__ w2,
    float* __restrict__ dbl1, float* __restrict__ dbl2)
{
    const int lane = threadIdx.x & 31;
    const int wid  = threadIdx.x >> 5;
    const int z = blockIdx.z;                  // s*8 + b*4 + k
    const int s = z >> 3, b = (z >> 2) & 1, k = z & 3;
    const float* x   = s ? x2 : x1;
    const float* w   = s ? w2 : w1;
    float*       dbl = s ? dbl2 : dbl1;

    const int tileL = blockIdx.x * 4 + wid;    // 0..195
    const int tileM = blockIdx.y;              // 0..2 (44 rows -> 3 tiles)
    const int m    = lane & 15;
    const int half = lane >> 4;
    const int lg   = tileL * 16 + (lane & 15); // output column l
    const int idx  = scan_idx(k, lg);          // cross-scan gather

    const int   mg  = tileM * 16 + m;          // row c (guard >= 44)
    const float msk = (mg < CC_) ? 1.f : 0.f;  // branch-free row mask
    const float* wrow = w + (size_t)(k * CC_ + (mg < CC_ ? mg : CC_ - 1)) * D_;
    const float* xcol = x + (size_t)b * D_ * L_ + idx;

    v8f acc = {};
    #pragma unroll
    for (int k0 = 0; k0 < D_; k0 += 32) {
        // A fragment: lane row m; elements = two runs of 8 contiguous floats
        //   a[0..7]  <- wrow[k0 + half*8 + 0..7]
        //   a[8..15] <- wrow[k0 + 16 + half*8 + 0..7]
        const float4 a0 = *(const float4*)(wrow + k0 + half * 8);
        const float4 a1 = *(const float4*)(wrow + k0 + half * 8 + 4);
        const float4 a2 = *(const float4*)(wrow + k0 + 16 + half * 8);
        const float4 a3 = *(const float4*)(wrow + k0 + 16 + half * 8 + 4);
        v16h a;
        a[0]  = (_Float16)(a0.x * msk); a[1]  = (_Float16)(a0.y * msk);
        a[2]  = (_Float16)(a0.z * msk); a[3]  = (_Float16)(a0.w * msk);
        a[4]  = (_Float16)(a1.x * msk); a[5]  = (_Float16)(a1.y * msk);
        a[6]  = (_Float16)(a1.z * msk); a[7]  = (_Float16)(a1.w * msk);
        a[8]  = (_Float16)(a2.x * msk); a[9]  = (_Float16)(a2.y * msk);
        a[10] = (_Float16)(a2.z * msk); a[11] = (_Float16)(a2.w * msk);
        a[12] = (_Float16)(a3.x * msk); a[13] = (_Float16)(a3.y * msk);
        a[14] = (_Float16)(a3.z * msk); a[15] = (_Float16)(a3.w * msk);
        // B fragment: lane column n; element t <- K = half*8 + t (t<8), 16 + half*8 + (t-8) (t>=8)
        v16h bm;
        #pragma unroll
        for (int t = 0; t < 16; ++t)
            bm[t] = (_Float16)xcol[(size_t)(k0 + ((t < 8) ? (half * 8 + t) : (16 + half * 8 + (t - 8)))) * L_];
        acc = __builtin_amdgcn_wmma_f32_16x16x32_f16(
                  false, a, false, bm, (short)0, acc, false, false);
    }
    #pragma unroll
    for (int v = 0; v < 8; ++v) {
        const int mo = tileM * 16 + v + 8 * half;   // C/D layout: M = v + 8*half
        if (mo < CC_)
            dbl[((size_t)(b * K_ + k) * CC_ + mo) * L_ + lg] = acc[v];
    }
}

// ---------------------------------------------------------------------------
// GEMM2: delta[b,k,d,l] = softplus( sum_r dtw[k,d,r]*dbl[b,k,r,l] + bias[k,d] )
// K-dim 12 padded to one 32-wide WMMA step; loads unconditional + masked.
__global__ __launch_bounds__(128) void k_proj_delta(
    const float* __restrict__ dtw1, const float* __restrict__ dtw2,
    const float* __restrict__ dtb1, const float* __restrict__ dtb2,
    const float* __restrict__ dbl1, const float* __restrict__ dbl2,
    float* __restrict__ del1, float* __restrict__ del2)
{
    const int lane = threadIdx.x & 31;
    const int wid  = threadIdx.x >> 5;
    const int z = blockIdx.z;
    const int s = z >> 3, b = (z >> 2) & 1, k = z & 3;
    const float* dtw = s ? dtw2 : dtw1;
    const float* dtb = s ? dtb2 : dtb1;
    const float* dbl = s ? dbl2 : dbl1;
    float*       del = s ? del2 : del1;

    const int tileL = blockIdx.x * 4 + wid;    // 0..195
    const int tileM = blockIdx.y;              // 0..11 (192 rows)
    const int m    = lane & 15;
    const int half = lane >> 4;
    const int lg   = tileL * 16 + (lane & 15);
    const int mg   = tileM * 16 + m;           // d row, always < 192

    const float* wrow = dtw + (size_t)(k * D_ + mg) * R_;
    const float* bcol = dbl + (size_t)(b * K_ + k) * CC_ * L_ + lg;

    v16h a, bm;
    #pragma unroll
    for (int e = 0; e < 16; ++e) {
        const int j = e >> 1, pr = e & 1;
        const int ka = ((j < 4) ? (half * 8 + 2 * j)
                                : (16 + half * 8 + 2 * (j - 4))) + pr;
        const float av = wrow[ka < R_ ? ka : 0];         // unconditional load
        a[e] = (_Float16)(ka < R_ ? av : 0.f);           // masked select
    }
    #pragma unroll
    for (int t = 0; t < 16; ++t) {
        const int kb = (t < 8) ? (half * 8 + t) : (16 + half * 8 + (t - 8));
        const float bv = bcol[(size_t)(kb < R_ ? kb : 0) * L_];
        bm[t] = (_Float16)(kb < R_ ? bv : 0.f);
    }
    v8f acc = {};
    acc = __builtin_amdgcn_wmma_f32_16x16x32_f16(
              false, a, false, bm, (short)0, acc, false, false);

    #pragma unroll
    for (int v = 0; v < 8; ++v) {
        const int mo = tileM * 16 + v + 8 * half;           // d
        float val = acc[v] + dtb[k * D_ + mo];
        val = fmaxf(val, 0.f) + log1pf(__expf(-fabsf(val))); // stable softplus
        del[((size_t)(b * K_ + k) * D_ + mo) * L_ + lg] = val;
    }
}

// ---------------------------------------------------------------------------
// Selective scan. lane = n + 16*(d&1): each lane owns one h[d,n] state.
// y_d = shfl_xor-reduce over 16 lanes; merged via atomicAdd at idx_k(l).
// Cross modality: stream s uses B from its own dbl, C from the OTHER stream.
__global__ __launch_bounds__(256) void k_scan(
    const float* __restrict__ x1,   const float* __restrict__ x2,
    const float* __restrict__ dbl1, const float* __restrict__ dbl2,
    const float* __restrict__ del1, const float* __restrict__ del2,
    const float* __restrict__ Al1,  const float* __restrict__ Al2,
    const float* __restrict__ Ds1,  const float* __restrict__ Ds2,
    float* __restrict__ ym1, float* __restrict__ ym2)
{
    const int lane = threadIdx.x & 31;
    const int wid  = threadIdx.x >> 5;
    const int z = blockIdx.z;                  // s*2 + b
    const int s = z >> 1, b = z & 1;
    const int k = blockIdx.y;
    const int d = blockIdx.x * 16 + wid * 2 + (lane >> 4);
    const int n = lane & 15;

    const float* x    = s ? x2   : x1;
    const float* del  = s ? del2 : del1;
    const float* dblB = s ? dbl2 : dbl1;
    const float* dblC = s ? dbl1 : dbl2;       // cross-modal C
    const float* Alog = s ? Al2  : Al1;
    const float* Dp   = s ? Ds2  : Ds1;
    float*       ym   = s ? ym2  : ym1;

    const float A  = -__expf(Alog[(size_t)(k * D_ + d) * N_ + n]);
    const float Dv = Dp[k * D_ + d];
    const float* drow = del  + ((size_t)(b * K_ + k) * D_  + d)           * L_;
    const float* Brow = dblB + ((size_t)(b * K_ + k) * CC_ + R_ + n)      * L_;
    const float* Crow = dblC + ((size_t)(b * K_ + k) * CC_ + R_ + N_ + n) * L_;
    const float* xrow = x  + (size_t)(b * D_ + d) * L_;
    float*       yrow = ym + (size_t)(b * D_ + d) * L_;

    float h = 0.f;
    for (int l = 0; l < L_; ++l) {
        const int   idx = scan_idx(k, l);      // uniform per block (k uniform)
        const float u   = xrow[idx];
        const float dt  = drow[l];
        const float Bv  = Brow[l];
        const float Cv  = Crow[l];
        const float dA  = __expf(dt * A);
        h = fmaf(dt * u, Bv, dA * h);
        float p = h * Cv;
        p += __shfl_xor(p, 1, 32);
        p += __shfl_xor(p, 2, 32);
        p += __shfl_xor(p, 4, 32);
        p += __shfl_xor(p, 8, 32);             // sum over 16-lane group
        if (n == 0)
            atomicAdd(&yrow[idx], p + Dv * u);
    }
}

// ---------------------------------------------------------------------------
// LayerNorm over D=192 per (s,b,l); one wave per position, 6 channels/lane.
__global__ __launch_bounds__(256) void k_merge_ln(
    const float* __restrict__ ym1, const float* __restrict__ ym2,
    const float* __restrict__ lnw, const float* __restrict__ lnb,
    float* __restrict__ out)
{
    const int lane = threadIdx.x & 31;
    const int wid  = threadIdx.x >> 5;
    const int p = blockIdx.x * 8 + wid;        // 0 .. 2*B*L-1
    const int s = p / (B_ * L_);
    const int rem = p - s * (B_ * L_);
    const int b = rem / L_, l = rem % L_;
    const float* ym = s ? ym2 : ym1;

    float v[6];
    float sum = 0.f;
    #pragma unroll
    for (int i = 0; i < 6; ++i) {
        const int d = lane + 32 * i;
        v[i] = ym[(size_t)(b * D_ + d) * L_ + l];
        sum += v[i];
    }
    #pragma unroll
    for (int mm = 16; mm >= 1; mm >>= 1) sum += __shfl_xor(sum, mm, 32);
    const float mean = sum * (1.f / D_);
    float var = 0.f;
    #pragma unroll
    for (int i = 0; i < 6; ++i) { const float t = v[i] - mean; var += t * t; }
    #pragma unroll
    for (int mm = 16; mm >= 1; mm >>= 1) var += __shfl_xor(var, mm, 32);
    const float rstd = rsqrtf(var * (1.f / D_) + 1e-5f);

    float* o = out + (size_t)s * B_ * L_ * D_ + ((size_t)b * L_ + l) * D_;
    #pragma unroll
    for (int i = 0; i < 6; ++i) {
        const int d = lane + 32 * i;
        o[d] = (v[i] - mean) * rstd * lnw[d] + lnb[d];
    }
}

// ---------------------------------------------------------------------------
extern "C" void kernel_launch(void* const* d_in, const int* in_sizes, int n_in,
                              void* d_out, int out_size, void* d_ws, size_t ws_size,
                              hipStream_t stream)
{
    (void)in_sizes; (void)n_in; (void)out_size; (void)ws_size;
    const float* x1   = (const float*)d_in[0];
    const float* x2   = (const float*)d_in[1];
    const float* w1   = (const float*)d_in[2];
    const float* w2   = (const float*)d_in[3];
    const float* dtw1 = (const float*)d_in[4];
    const float* dtw2 = (const float*)d_in[5];
    const float* dtb1 = (const float*)d_in[6];
    const float* dtb2 = (const float*)d_in[7];
    const float* Al1  = (const float*)d_in[8];
    const float* Al2  = (const float*)d_in[9];
    const float* Ds1  = (const float*)d_in[10];
    const float* Ds2  = (const float*)d_in[11];
    const float* lnw  = (const float*)d_in[12];
    const float* lnb  = (const float*)d_in[13];
    float* out = (float*)d_out;

    // workspace layout (floats): dbl1|dbl2|del1|del2|ym1|ym2  (~54.4 MB)
    const size_t DBL = (size_t)B_ * K_ * CC_ * L_;   // 1,103,872
    const size_t DEL = (size_t)B_ * K_ * D_  * L_;   // 4,816,896
    const size_t YM  = (size_t)B_ * D_ * L_;         // 1,204,224
    float* ws   = (float*)d_ws;
    float* dbl1 = ws;
    float* dbl2 = dbl1 + DBL;
    float* del1 = dbl2 + DBL;
    float* del2 = del1 + DEL;
    float* ym1  = del2 + DEL;
    float* ym2  = ym1  + YM;

    // 1) zero the merge accumulators (ym1, ym2 are contiguous)
    {
        const int n = (int)(2 * YM);
        k_zero<<<(n + 255) / 256, 256, 0, stream>>>(ym1, n);
    }
    // 2) dbl projection GEMM (WMMA): 196 L-tiles, 3 M-tiles, 16 (s,b,k)
    k_proj_dbl<<<dim3(49, 3, 16), 128, 0, stream>>>(x1, x2, w1, w2, dbl1, dbl2);
    // 3) dt projection GEMM (WMMA) + bias + softplus: 12 M-tiles
    k_proj_delta<<<dim3(49, 12, 16), 128, 0, stream>>>(dtw1, dtw2, dtb1, dtb2,
                                                       dbl1, dbl2, del1, del2);
    // 4) sequential selective scan + cross-merge via atomicAdd
    k_scan<<<dim3(D_ / 16, K_, 4), 256, 0, stream>>>(x1, x2, dbl1, dbl2,
                                                     del1, del2, Al1, Al2,
                                                     Ds1, Ds2, ym1, ym2);
    // 5) layernorm -> output [y1 | y2], each [B,H,W,D]
    k_merge_ln<<<(2 * B_ * L_) / 8, 256, 0, stream>>>(ym1, ym2, lnw, lnb, out);
}
